// LeViT_53042846106117
// MI455X (gfx1250) — matrix operand order repaired and checked by reading, not compiled, not run on hardware
//
#include <hip/hip_runtime.h>
#include <stdint.h>

// ---------------- problem constants ----------------
#define TOK    196          // N = 14*14 tokens
#define NPAD   224          // tokens padded to 7 * 32
#define HEADS  12
#define KD     32           // key dim
#define DV     128          // value dim per head
#define DIM_   384
#define HQKV   2304         // qkv output channels
#define DH_    1536         // HEADS * DV
#define BATCH  256
#define ROWS   (BATCH * TOK)     // 50176 = 32 * 1568
#define BN_EPS 1e-5f
#define QSCALE 0.17677669529663687f   // 1/sqrt(32), folded into Q weights

typedef __attribute__((ext_vector_type(16))) __bf16 v16bf;
typedef __attribute__((ext_vector_type(8)))  __bf16 v8bf;
typedef __attribute__((ext_vector_type(8)))  float  v8f;

// ---------------- helpers ----------------
static __device__ __forceinline__ __bf16 f2bf(float f) {
  union { float f; uint32_t u; } a; a.f = f;
  uint32_t r = a.u + 0x7FFFu + ((a.u >> 16) & 1u);  // RNE-ish
  union { uint16_t s; __bf16 b; } o; o.s = (uint16_t)(r >> 16);
  return o.b;
}

// A fragment 16x32 bf16 (MxK): lane = m + 16*half holds K in
// {half*8 .. half*8+7} U {16+half*8 .. 16+half*8+7}  (ISA 7.12.2)
static __device__ __forceinline__ v16bf load_afrag(const __bf16* base, int stride, int lane) {
  int m = lane & 15, half = lane >> 4;
  const __bf16* p = base + (size_t)m * stride + half * 8;
  union { v16bf v; v8bf h[2]; } u;
  u.h[0] = *(const v8bf*)(p);
  u.h[1] = *(const v8bf*)(p + 16);
  return u.v;
}

// B fragment 32x16 bf16 (KxN): lane = n + 16*half holds K in
// {half*8 .. half*8+7} U {16+half*8 .. 16+half*8+7}
// of column n; base points at B[0][0] where per-lane column steps by `stride`.
static __device__ __forceinline__ v16bf load_bfrag(const __bf16* base, int stride, int lane) {
  int n = lane & 15, half = lane >> 4;
  const __bf16* p = base + (size_t)n * stride + half * 8;
  union { v16bf v; v8bf h[2]; } u;
  u.h[0] = *(const v8bf*)(p);
  u.h[1] = *(const v8bf*)(p + 16);
  return u.v;
}

static __device__ __forceinline__ v8f wmma_bf16(v16bf a, v16bf b, v8f c) {
  return __builtin_amdgcn_wmma_f32_16x16x32_bf16(false, a, false, b, (short)0, c, false, false);
}

// CDNA5 async global->LDS copy, 16B per lane (ASYNCcnt tracked).
// Flat-LDS addressing uses addr[31:0] as the LDS offset, so the low 32 bits
// of a generic __shared__ pointer are a valid VDST LDS address.
static __device__ __forceinline__ void async_copy16(const void* gptr, void* lptr) {
  asm volatile("global_load_async_to_lds_b128 %0, %1, off"
               :: "v"((uint32_t)(uintptr_t)lptr), "v"(gptr)
               : "memory");
}
static __device__ __forceinline__ void wait_async0() {
  asm volatile("s_wait_asynccnt 0x0" ::: "memory");
}

// ---------------- kernel 1: convert x -> bf16 ----------------
__global__ __launch_bounds__(256) void cvt_x_kernel(const float* __restrict__ x,
                                                    __bf16* __restrict__ xb, int n) {
  int stride = gridDim.x * blockDim.x;
  for (int i = blockIdx.x * blockDim.x + threadIdx.x; i < n; i += stride)
    xb[i] = f2bf(x[i]);
}

// ---------------- kernel 2: fold BN into bf16 weights ----------------
__global__ __launch_bounds__(256) void prep_w_kernel(
    const float* __restrict__ qkv_w, const float* __restrict__ g,
    const float* __restrict__ be, const float* __restrict__ mu, const float* __restrict__ var,
    const float* __restrict__ proj_w, const float* __restrict__ pg,
    const float* __restrict__ pbe, const float* __restrict__ pmu, const float* __restrict__ pvar,
    __bf16* __restrict__ wb, float* __restrict__ qb,
    __bf16* __restrict__ pw, float* __restrict__ pb) {
  int stride = gridDim.x * blockDim.x;
  int tid = blockIdx.x * blockDim.x + threadIdx.x;
  for (int i = tid; i < HQKV * DIM_; i += stride) {
    int h = i / DIM_;
    float s = g[h] * rsqrtf(var[h] + BN_EPS);
    float f = ((h % 192) < KD) ? QSCALE : 1.f;      // fold attn scale into Q rows
    wb[i] = f2bf(qkv_w[i] * s * f);
  }
  for (int i = tid; i < HQKV; i += stride) {
    float s = g[i] * rsqrtf(var[i] + BN_EPS);
    float f = ((i % 192) < KD) ? QSCALE : 1.f;
    qb[i] = (be[i] - mu[i] * s) * f;
  }
  for (int i = tid; i < DIM_ * DH_; i += stride) {
    int c = i / DH_;
    float s = pg[c] * rsqrtf(pvar[c] + BN_EPS);
    pw[i] = f2bf(proj_w[i] * s);
  }
  for (int i = tid; i < DIM_; i += stride) {
    float s = pg[i] * rsqrtf(pvar[i] + BN_EPS);
    pb[i] = pbe[i] - pmu[i] * s;
  }
}

// ---------------- kernel 3: zero the token-padding regions ----------------
__global__ __launch_bounds__(256) void zero_pads_kernel(__bf16* __restrict__ Q,
                                                        __bf16* __restrict__ K,
                                                        __bf16* __restrict__ Vt) {
  int bh = blockIdx.x;                       // BATCH*HEADS blocks
  union { uint16_t s; __bf16 b; } z; z.s = 0;
  size_t qkBase = (size_t)bh * NPAD * KD + (size_t)TOK * KD;
  for (int i = threadIdx.x; i < (NPAD - TOK) * KD; i += blockDim.x) {
    Q[qkBase + i] = z.b;
    K[qkBase + i] = z.b;
  }
  size_t vBase = (size_t)bh * DV * NPAD;
  for (int i = threadIdx.x; i < DV * (NPAD - TOK); i += blockDim.x) {
    int d = i / (NPAD - TOK), c = TOK + i % (NPAD - TOK);
    Vt[vBase + (size_t)d * NPAD + c] = z.b;
  }
}

// ---------------- kernel 4: QKV GEMM (bf16 WMMA) + BN epilogue + layout ----
// xb[ROWS][384] @ wb[2304][384]^T -> route into Q/K/Vt bf16 buffers.
// wave = 32x64 tile (2 A-frags reuse each B-frag); grid = (ROWS/32=1568) x 36.
__global__ __launch_bounds__(256) void qkv_kernel(
    const __bf16* __restrict__ xb, const __bf16* __restrict__ wb,
    const float* __restrict__ qb,
    __bf16* __restrict__ Q, __bf16* __restrict__ K, __bf16* __restrict__ Vt) {
  int gw = blockIdx.x * 8 + (threadIdx.x >> 5);
  int lane = threadIdx.x & 31;
  int rowTile = gw / 36, colGrp = gw % 36;
  int rowbase = rowTile * 32;
  int colbase = colGrp * 64;

  v8f acc0[4], acc1[4];
#pragma unroll
  for (int t = 0; t < 4; ++t) {
    acc0[t] = (v8f){0.f,0.f,0.f,0.f,0.f,0.f,0.f,0.f};
    acc1[t] = (v8f){0.f,0.f,0.f,0.f,0.f,0.f,0.f,0.f};
  }

  const __bf16* arow0 = xb + (size_t)rowbase * DIM_;
  const __bf16* arow1 = arow0 + (size_t)16 * DIM_;
#pragma unroll 4
  for (int k = 0; k < DIM_; k += 32) {
    __builtin_prefetch(arow0 + k + 64, 0, 3);
    __builtin_prefetch(arow1 + k + 64, 0, 3);
    v16bf a0 = load_afrag(arow0 + k, DIM_, lane);
    v16bf a1 = load_afrag(arow1 + k, DIM_, lane);
#pragma unroll
    for (int t = 0; t < 4; ++t) {
      v16bf b = load_bfrag(wb + (size_t)(colbase + t * 16) * DIM_ + k, DIM_, lane);
      acc0[t] = wmma_bf16(a0, b, acc0[t]);
      acc1[t] = wmma_bf16(a1, b, acc1[t]);
    }
  }

  int lhalf = lane >> 4, lmod = lane & 15;
#pragma unroll
  for (int t = 0; t < 4; ++t) {
    int c = colbase + t * 16 + lmod;          // output channel (fixed per lane)
    float bias = qb[c];
    int hd = c / 192, r = c % 192;
#pragma unroll
    for (int sub = 0; sub < 2; ++sub) {
#pragma unroll
      for (int rr = 0; rr < 8; ++rr) {
        int row = rowbase + sub * 16 + 8 * lhalf + rr;
        int b = row / TOK, n = row % TOK;
        float av = sub ? acc1[t][rr] : acc0[t][rr];
        __bf16 v = f2bf(av + bias);
        size_t bh = (size_t)(b * HEADS + hd);
        if (r < KD)            Q[bh * NPAD * KD + (size_t)n * KD + r] = v;
        else if (r < 2 * KD)   K[bh * NPAD * KD + (size_t)n * KD + (r - KD)] = v;
        else                   Vt[bh * DV * NPAD + (size_t)(r - 2 * KD) * NPAD + n] = v;
      }
    }
  }
}

// ---------------- kernel 5: fused attention (flash-style, WMMA) ----------
// one block per (b,h); 13 waves, one 16-row tile each; K-blocks of 32 cols.
// K and V^T staged once into LDS (WGP has 320 KB) via async global->LDS.
__global__ __launch_bounds__(416) void attn_kernel(
    const __bf16* __restrict__ Q, const __bf16* __restrict__ Km,
    const __bf16* __restrict__ Vt, const float* __restrict__ ab, int n_off,
    const int* __restrict__ bidx, __bf16* __restrict__ outb) {
  __shared__ float s_ab[TOK];
  __shared__ __align__(32) __bf16 s_k[NPAD][KD];        // 14 KB
  __shared__ __align__(32) __bf16 s_v[DV][NPAD];        // 57 KB
  __shared__ __align__(32) __bf16 s_p[13][16][32];      // 13 KB

  int bh = blockIdx.x;
  int b = bh / HEADS, h = bh % HEADS;
  int tid = threadIdx.x;

  const __bf16* Qb = Q + (size_t)bh * NPAD * KD;
  const __bf16* Kb = Km + (size_t)bh * NPAD * KD;
  const __bf16* Vb = Vt + (size_t)bh * DV * NPAD;

  // async stage: K (224x32) and V^T (128x224) -> LDS, 16B per lane per op
  {
    const v8bf* ks = (const v8bf*)Kb;
    v8bf* kd = (v8bf*)&s_k[0][0];
    for (int i = tid; i < NPAD * KD / 8; i += 416) async_copy16(ks + i, kd + i);
    const v8bf* vs = (const v8bf*)Vb;
    v8bf* vd = (v8bf*)&s_v[0][0];
    for (int i = tid; i < DV * NPAD / 8; i += 416) async_copy16(vs + i, vd + i);
  }
  if (tid < n_off && tid < TOK) s_ab[tid] = ab[h * n_off + tid];  // overlaps async
  wait_async0();               // drain this wave's ASYNCcnt before the barrier
  __syncthreads();

  int w = tid >> 5;              // wave id = row tile 0..12
  int lane = tid & 31;
  int lhalf = lane >> 4, lmod = lane & 15;
  int rowbase = w * 16;

  v16bf aq = load_afrag(Qb + (size_t)rowbase * KD, KD, lane);  // K=32: single step

  v8f O[8];
#pragma unroll
  for (int t = 0; t < 8; ++t) O[t] = (v8f){0.f,0.f,0.f,0.f,0.f,0.f,0.f,0.f};
  float mrow[8], lrow[8];
#pragma unroll
  for (int r = 0; r < 8; ++r) { mrow[r] = -1e30f; lrow[r] = 0.f; }

  for (int jp = 0; jp < 7; ++jp) {
    int col0 = jp * 32;
    // --- bias into C fragments (C layout: M = 8*lhalf+r, N = lmod) ---
    v8f c0, c1;
#pragma unroll
    for (int r = 0; r < 8; ++r) {
      int m = rowbase + 8 * lhalf + r;  int mi = m < TOK ? m : TOK - 1;
      int n0 = col0 + lmod;             int n0i = n0 < TOK ? n0 : TOK - 1;
      int n1 = col0 + 16 + lmod;        int n1i = n1 < TOK ? n1 : TOK - 1;
      c0[r] = s_ab[bidx[mi * TOK + n0i]];
      c1[r] = s_ab[bidx[mi * TOK + n1i]];
    }
    // --- S = Q K^T + bias (two 16x16 tiles), K fragments from LDS ---
    v16bf bk0 = load_bfrag(&s_k[col0][0], KD, lane);
    v16bf bk1 = load_bfrag(&s_k[col0 + 16][0], KD, lane);
    v8f S0 = wmma_bf16(aq, bk0, c0);
    v8f S1 = wmma_bf16(aq, bk1, c1);
    bool v0 = (col0 + lmod) < TOK;
    bool v1 = (col0 + 16 + lmod) < TOK;
#pragma unroll
    for (int r = 0; r < 8; ++r) {
      S0[r] = v0 ? S0[r] : -1e30f;
      S1[r] = v1 ? S1[r] : -1e30f;
    }
    // --- online softmax update ---
    float mnew[8], scale[8];
#pragma unroll
    for (int r = 0; r < 8; ++r) {
      float t = fmaxf(S0[r], S1[r]);
      t = fmaxf(t, __shfl_xor(t, 1, 32));
      t = fmaxf(t, __shfl_xor(t, 2, 32));
      t = fmaxf(t, __shfl_xor(t, 4, 32));
      t = fmaxf(t, __shfl_xor(t, 8, 32));
      mnew[r] = fmaxf(mrow[r], t);
      scale[r] = __expf(mrow[r] - mnew[r]);
      mrow[r] = mnew[r];
    }
#pragma unroll
    for (int r = 0; r < 8; ++r) {
      float p0 = __expf(S0[r] - mnew[r]);  p0 = v0 ? p0 : 0.f;
      float p1 = __expf(S1[r] - mnew[r]);  p1 = v1 ? p1 : 0.f;
      float sl = p0 + p1;
      sl += __shfl_xor(sl, 1, 32);
      sl += __shfl_xor(sl, 2, 32);
      sl += __shfl_xor(sl, 4, 32);
      sl += __shfl_xor(sl, 8, 32);
      lrow[r] = lrow[r] * scale[r] + sl;
      S0[r] = p0; S1[r] = p1;
    }
#pragma unroll
    for (int t = 0; t < 8; ++t)
#pragma unroll
      for (int r = 0; r < 8; ++r) O[t][r] *= scale[r];
    // --- stage P through LDS: C layout -> A fragment (16x32 bf16) ---
#pragma unroll
    for (int r = 0; r < 8; ++r) {
      int m = 8 * lhalf + r;
      s_p[w][m][lmod]      = f2bf(S0[r]);
      s_p[w][m][16 + lmod] = f2bf(S1[r]);
    }
    v16bf ap = load_afrag(&s_p[w][0][0], 32, lane);   // same-wave LDS: DS in-order
    // --- O += P @ V  (8 d-tiles of 16), V fragments from LDS ---
#pragma unroll
    for (int t = 0; t < 8; ++t) {
      v16bf bv = load_bfrag(&s_v[t * 16][col0], NPAD, lane);
      O[t] = wmma_bf16(ap, bv, O[t]);
    }
  }

  // --- epilogue: 1/l, hardswish, store bf16 into [b][n][h*128+d] ---
  float rl[8];
#pragma unroll
  for (int r = 0; r < 8; ++r) rl[r] = 1.f / lrow[r];
#pragma unroll
  for (int t = 0; t < 8; ++t) {
#pragma unroll
    for (int r = 0; r < 8; ++r) {
      int m = rowbase + 8 * lhalf + r;
      if (m < TOK) {
        float o = O[t][r] * rl[r];
        float hs = o * fminf(fmaxf(o + 3.f, 0.f), 6.f) * (1.f / 6.f);
        outb[((size_t)b * TOK + m) * DH_ + h * DV + t * 16 + lmod] = f2bf(hs);
      }
    }
  }
}

// ---------------- kernel 6: proj GEMM (bf16 WMMA) + BN, f32 out ----------
// wave = 32x64 tile; grid = (ROWS/32=1568) x 6 colgroups.
__global__ __launch_bounds__(256) void proj_kernel(
    const __bf16* __restrict__ ob, const __bf16* __restrict__ pw,
    const float* __restrict__ pb, float* __restrict__ y) {
  int gw = blockIdx.x * 8 + (threadIdx.x >> 5);
  int lane = threadIdx.x & 31;
  int rowTile = gw / 6, colGrp = gw % 6;
  int rowbase = rowTile * 32, colbase = colGrp * 64;

  v8f acc0[4], acc1[4];
#pragma unroll
  for (int t = 0; t < 4; ++t) {
    acc0[t] = (v8f){0.f,0.f,0.f,0.f,0.f,0.f,0.f,0.f};
    acc1[t] = (v8f){0.f,0.f,0.f,0.f,0.f,0.f,0.f,0.f};
  }

  const __bf16* arow0 = ob + (size_t)rowbase * DH_;
  const __bf16* arow1 = arow0 + (size_t)16 * DH_;
#pragma unroll 4
  for (int k = 0; k < DH_; k += 32) {
    __builtin_prefetch(arow0 + k + 64, 0, 3);
    __builtin_prefetch(arow1 + k + 64, 0, 3);
    v16bf a0 = load_afrag(arow0 + k, DH_, lane);
    v16bf a1 = load_afrag(arow1 + k, DH_, lane);
#pragma unroll
    for (int t = 0; t < 4; ++t) {
      v16bf b = load_bfrag(pw + (size_t)(colbase + t * 16) * DH_ + k, DH_, lane);
      acc0[t] = wmma_bf16(a0, b, acc0[t]);
      acc1[t] = wmma_bf16(a1, b, acc1[t]);
    }
  }
  int lhalf = lane >> 4, lmod = lane & 15;
#pragma unroll
  for (int t = 0; t < 4; ++t) {
    int c = colbase + t * 16 + lmod;
    float bias = pb[c];
#pragma unroll
    for (int rr = 0; rr < 8; ++rr) {
      int row0 = rowbase + 8 * lhalf + rr;
      y[(size_t)row0 * DIM_ + c] = acc0[t][rr] + bias;
      y[(size_t)(row0 + 16) * DIM_ + c] = acc1[t][rr] + bias;
    }
  }
}

// ---------------- host launcher ----------------
extern "C" void kernel_launch(void* const* d_in, const int* in_sizes, int n_in,
                              void* d_out, int out_size, void* d_ws, size_t ws_size,
                              hipStream_t stream) {
  const float* x      = (const float*)d_in[0];
  const float* qkv_w  = (const float*)d_in[1];
  const float* qg     = (const float*)d_in[2];
  const float* qbe    = (const float*)d_in[3];
  const float* qmu    = (const float*)d_in[4];
  const float* qvar   = (const float*)d_in[5];
  const float* ab     = (const float*)d_in[6];
  const float* proj_w = (const float*)d_in[7];
  const float* pg     = (const float*)d_in[8];
  const float* pbe    = (const float*)d_in[9];
  const float* pmu    = (const float*)d_in[10];
  const float* pvar   = (const float*)d_in[11];
  const int*   bidx   = (const int*)d_in[12];
  float* y = (float*)d_out;
  int n_off = in_sizes[6] / HEADS;

  // workspace carve (all offsets 256B-aligned); total ~460 MB
  size_t off = 0;
  auto alloc = [&](size_t bytes) {
    void* p = (char*)d_ws + off;
    off += (bytes + 255) & ~(size_t)255;
    return p;
  };
  __bf16* xb  = (__bf16*)alloc((size_t)ROWS * DIM_ * 2);
  __bf16* wb  = (__bf16*)alloc((size_t)HQKV * DIM_ * 2);
  float*  qb  = (float*) alloc((size_t)HQKV * 4);
  __bf16* pw  = (__bf16*)alloc((size_t)DIM_ * DH_ * 2);
  float*  pb  = (float*) alloc((size_t)DIM_ * 4);
  __bf16* Qb  = (__bf16*)alloc((size_t)BATCH * HEADS * NPAD * KD * 2);
  __bf16* Kb  = (__bf16*)alloc((size_t)BATCH * HEADS * NPAD * KD * 2);
  __bf16* Vt  = (__bf16*)alloc((size_t)BATCH * HEADS * DV * NPAD * 2);
  __bf16* ob  = (__bf16*)alloc((size_t)ROWS * DH_ * 2);
  (void)ws_size;

  cvt_x_kernel<<<4096, 256, 0, stream>>>(x, xb, ROWS * DIM_);
  prep_w_kernel<<<2048, 256, 0, stream>>>(qkv_w, qg, qbe, qmu, qvar,
                                          proj_w, pg, pbe, pmu, pvar,
                                          wb, qb, pw, pb);
  zero_pads_kernel<<<BATCH * HEADS, 256, 0, stream>>>(Qb, Kb, Vt);
  // QKV: 1568 row tiles * 36 col groups = 56448 waves / 8 = 7056 blocks
  qkv_kernel<<<7056, 256, 0, stream>>>(xb, wb, qb, Qb, Kb, Vt);
  // attention: one block per (batch, head)
  attn_kernel<<<BATCH * HEADS, 416, 0, stream>>>(Qb, Kb, Vt, ab, n_off, bidx, ob);
  // proj: 1568 row tiles * 6 col groups = 9408 waves / 8 = 1176 blocks
  proj_kernel<<<1176, 256, 0, stream>>>(ob, pw, pb, y);
}
